// WindowAttention_18382460027445
// MI455X (gfx1250) — compile-verified, matrix operand order checked by reading, not run on hardware
//
#include <hip/hip_runtime.h>
#include <hip/hip_bf16.h>

// ---- problem constants ----
#define N_TOK 49
#define DIM_  128
#define H_    4
#define HD_   32
#define NW_   64
#define NPAD  64   // tokens padded to 4 WMMA tiles

typedef _Float16 h16 __attribute__((ext_vector_type(16)));
typedef _Float16 h8  __attribute__((ext_vector_type(8)));
typedef _Float16 h4  __attribute__((ext_vector_type(4)));
typedef float    f8  __attribute__((ext_vector_type(8)));
typedef float    f4  __attribute__((ext_vector_type(4)));

union H16U { h16 v; h8 h[2]; _Float16 e[16]; };

// K index covered by fragment element e for lane-half kh (ISA 05_wmma 16-bit A/B layout):
// e 0..7  -> K = 8*kh + e ; e 8..15 -> K = 16 + 8*kh + (e-8)
__device__ __forceinline__ int kmap_elem(int e, int kh) {
  return (e < 8) ? (8 * kh + e) : (16 + 8 * kh + (e - 8));
}

// ---------------- prep kernels: pre-swizzle weights into B-fragment order ----------------
__global__ void prep_wq(const float* __restrict__ w, _Float16* __restrict__ dst) {
  int idx = blockIdx.x * 256 + threadIdx.x;
  if (idx >= 4 * 24 * 32 * 16) return;            // kt(4) x nt(24) x lane(32) x elem(16)
  int e = idx & 15;
  int lane = (idx >> 4) & 31;
  int tile = idx >> 9;
  int nt = tile % 24, kt = tile / 24;
  int K = kt * 32 + kmap_elem(e, lane >> 4);
  int n = nt * 16 + (lane & 15);
  dst[idx] = (_Float16)w[K * (3 * DIM_) + n];
}

__global__ void prep_wp(const float* __restrict__ w, _Float16* __restrict__ dst) {
  int idx = blockIdx.x * 256 + threadIdx.x;
  if (idx >= 4 * 8 * 32 * 16) return;             // kt(4) x nt(8) x lane(32) x elem(16)
  int e = idx & 15;
  int lane = (idx >> 4) & 31;
  int tile = idx >> 9;
  int nt = tile % 8, kt = tile / 8;
  int K = kt * 32 + kmap_elem(e, lane >> 4);
  int n = nt * 16 + (lane & 15);
  dst[idx] = (_Float16)w[K * DIM_ + n];
}

__global__ void prep_rb(const float* __restrict__ table, const int* __restrict__ rel,
                        float* __restrict__ dst) {
  int idx = blockIdx.x * 256 + threadIdx.x;
  if (idx >= H_ * N_TOK * N_TOK) return;          // dst[h][n][m]
  int h = idx / (N_TOK * N_TOK);
  int rem = idx % (N_TOK * N_TOK);
  dst[idx] = table[rel[rem] * H_ + h];
}

// ---------------- fused window-attention kernel: 1 block = 1 window, 8 waves ----------------
// LDS (122880 B total -> 2 blocks / 16 waves per 320KB WGP):
//   qkv [64][392] f16  (50176 B) @ 0
//   r0 = xs [64][136] f16 -> later ao [64][136] f16 (17408 B) @ 50176
//   pb  [4][64][72] f16 (36864 B) @ 67584   (scores then probs, softmax in-place)
//   vt  [4][32][72] f16 (18432 B) @ 104448
__global__ __launch_bounds__(256) void swin_win_attn(
    const float* __restrict__ x, const float* __restrict__ mask,
    const float* __restrict__ b_qkv, const float* __restrict__ b_proj,
    const _Float16* __restrict__ wq, const _Float16* __restrict__ wp,
    const float* __restrict__ rb, float* __restrict__ out) {
  extern __shared__ char smem[];
  const int SQ = 392, SX = 136, SA = 136, SP = 72, SV = 72;

  _Float16* qkv = (_Float16*)smem;
  _Float16* xs  = (_Float16*)(smem + 50176);
  _Float16* ao  = (_Float16*)(smem + 50176);
  _Float16* pb  = (_Float16*)(smem + 67584);
  _Float16* vt  = (_Float16*)(smem + 104448);

  const int tid  = threadIdx.x;
  const int lane = tid & 31;
  // force wave index scalar so tile indices (mt/nt/h) are provably wave-uniform
  const int wave = __builtin_amdgcn_readfirstlane(tid >> 5);
  const int l15  = lane & 15;
  const int kh   = lane >> 4;
  const int w    = blockIdx.x;
  const float scale = 0.17677669529663687f;  // HD^-0.5

  // ---- phase 1: x tile -> LDS f16 (vectorized b128 loads / b64 ds stores) ----
  for (int idx = tid; idx < NPAD * DIM_ / 4; idx += 256) {
    int row = idx >> 5;          // wave-uniform (32 float4 per row)
    int c4  = idx & 31;
    f4 v = {};
    if (row < N_TOK) v = *(const f4*)(x + (size_t)w * N_TOK * DIM_ + row * DIM_ + c4 * 4);
    h4 hv; hv[0] = (_Float16)v[0]; hv[1] = (_Float16)v[1];
    hv[2] = (_Float16)v[2]; hv[3] = (_Float16)v[3];
    *(h4*)(xs + row * SX + c4 * 4) = hv;
  }
  __syncthreads();

  // ---- phase 2: qkv = x @ Wqkv + b ; fold scale into q ; store v also transposed ----
  for (int t = wave; t < 96; t += 8) {          // 4 Mtiles x 24 Ntiles (scalar loop)
    int mt = t / 24, nt = t % 24;
    f8 c = {};
    const _Float16* arow = xs + (mt * 16 + l15) * SX;
#pragma unroll
    for (int kt = 0; kt < 4; ++kt) {
      H16U a, b;
      a.h[0] = *(const h8*)(arow + kt * 32 + 8 * kh);
      a.h[1] = *(const h8*)(arow + kt * 32 + 16 + 8 * kh);
      const _Float16* bp = wq + ((size_t)(kt * 24 + nt) * 32 + lane) * 16;
      b.h[0] = *(const h8*)(bp);
      b.h[1] = *(const h8*)(bp + 8);
      c = __builtin_amdgcn_wmma_f32_16x16x32_f16(false, a.v, false, b.v, (short)0, c, false, false);
    }
    int n = nt * 16 + l15;
    float bias = b_qkv[n];
    float sc = (nt < 8) ? scale : 1.f;          // uniform select (q tiles are nt<8)
#pragma unroll
    for (int r = 0; r < 8; ++r) {
      int m = mt * 16 + r + 8 * kh;             // C/D layout: VGPR r -> row, lane-half -> +8
      float v = (c[r] + bias) * sc;
      qkv[m * SQ + n] = (_Float16)v;
    }
    if (nt >= 16) {                             // uniform: v tiles -> also store transposed
      int hh = (nt - 16) >> 1;
      int d  = ((nt - 16) & 1) * 16 + l15;
#pragma unroll
      for (int r = 0; r < 8; ++r) {
        int m = mt * 16 + r + 8 * kh;
        float v = c[r] + bias;
        vt[hh * 32 * SV + d * SV + m] = (_Float16)v;
      }
    }
  }
  __syncthreads();

  // ---- phase 3: S = q @ k^T + rel_bias + mask -> f16 into pb (K = HD -> 1 WMMA/tile) ----
  const float* mk = mask + (size_t)(w & (NW_ - 1)) * N_TOK * N_TOK;
  for (int t = wave; t < 64; t += 8) {          // 4 heads x 4 Mtiles x 4 Ntiles
    int h = t >> 4, rem = t & 15, mt = rem >> 2, nt = rem & 3;
    H16U a, b;
    const _Float16* arow = qkv + (mt * 16 + l15) * SQ + h * HD_;          // q
    a.h[0] = *(const h8*)(arow + 8 * kh);
    a.h[1] = *(const h8*)(arow + 16 + 8 * kh);
    const _Float16* brow = qkv + (nt * 16 + l15) * SQ + DIM_ + h * HD_;   // k (B = k^T)
    b.h[0] = *(const h8*)(brow + 8 * kh);
    b.h[1] = *(const h8*)(brow + 16 + 8 * kh);
    f8 c = {};
    c = __builtin_amdgcn_wmma_f32_16x16x32_f16(false, a.v, false, b.v, (short)0, c, false, false);
    int mcol = nt * 16 + l15;                   // key token
    int mc = (mcol < N_TOK) ? mcol : (N_TOK - 1);   // clamp -> always-safe loads
#pragma unroll
    for (int r = 0; r < 8; ++r) {
      int nrow = mt * 16 + r + 8 * kh;          // query token
      int nr = (nrow < N_TOK) ? nrow : (N_TOK - 1);
      float add = rb[(h * N_TOK + nr) * N_TOK + mc] + mk[nr * N_TOK + mc];
      float v = c[r] + ((nrow < N_TOK) ? add : 0.f);
      v = (mcol < N_TOK) ? v : -60000.f;        // padded keys -> exp() == 0
      pb[(h * NPAD + nrow) * SP + mcol] = (_Float16)v;
    }
  }
  __syncthreads();

  // ---- phase 3b: softmax in-place on pb, one (head,row) per thread, f32 math ----
  {
    int h = tid >> 6, row = tid & 63;
    _Float16* prow = pb + (h * NPAD + row) * SP;
    float mx = -3.4e38f;
    for (int i = 0; i < NPAD; ++i) mx = fmaxf(mx, (float)prow[i]);
    float sum = 0.f;
    for (int i = 0; i < NPAD; ++i) {
      float e = __expf((float)prow[i] - mx);
      sum += e;
      prow[i] = (_Float16)e;
    }
    float inv = 1.f / sum;
    for (int i = 0; i < NPAD; ++i) prow[i] = (_Float16)((float)prow[i] * inv);
  }
  __syncthreads();

  // ---- phase 4: out_h = P @ v  (B fragments contiguous thanks to vt) ----
  for (int t = wave; t < 32; t += 8) {          // 4 heads x 4 Mtiles x 2 Ntiles
    int h = t >> 3, rem = t & 7, mt = rem >> 1, nt = rem & 1;
    f8 c = {};
    const _Float16* arow = pb + (h * NPAD + mt * 16 + l15) * SP;
    const _Float16* brow = vt + h * 32 * SV + (nt * 16 + l15) * SV;
#pragma unroll
    for (int kt = 0; kt < 2; ++kt) {
      H16U a, b;
      a.h[0] = *(const h8*)(arow + kt * 32 + 8 * kh);
      a.h[1] = *(const h8*)(arow + kt * 32 + 16 + 8 * kh);
      b.h[0] = *(const h8*)(brow + kt * 32 + 8 * kh);
      b.h[1] = *(const h8*)(brow + kt * 32 + 16 + 8 * kh);
      c = __builtin_amdgcn_wmma_f32_16x16x32_f16(false, a.v, false, b.v, (short)0, c, false, false);
    }
    int dcol = h * HD_ + nt * 16 + l15;
#pragma unroll
    for (int r = 0; r < 8; ++r) {
      int m = mt * 16 + r + 8 * kh;
      ao[m * SA + dcol] = (_Float16)c[r];       // region0 (xs dead)
    }
  }
  __syncthreads();

  // ---- phase 5: out = ao @ Wproj + b_proj, store f32 rows 0..48 ----
  for (int t = wave; t < 32; t += 8) {          // 4 Mtiles x 8 Ntiles
    int mt = t >> 3, nt = t & 7;
    f8 c = {};
    const _Float16* arow = ao + (mt * 16 + l15) * SA;
#pragma unroll
    for (int kt = 0; kt < 4; ++kt) {
      H16U a, b;
      a.h[0] = *(const h8*)(arow + kt * 32 + 8 * kh);
      a.h[1] = *(const h8*)(arow + kt * 32 + 16 + 8 * kh);
      const _Float16* bp = wp + ((size_t)(kt * 8 + nt) * 32 + lane) * 16;
      b.h[0] = *(const h8*)bp;
      b.h[1] = *(const h8*)(bp + 8);
      c = __builtin_amdgcn_wmma_f32_16x16x32_f16(false, a.v, false, b.v, (short)0, c, false, false);
    }
    int n = nt * 16 + l15;
    float bias = b_proj[n];
#pragma unroll
    for (int r = 0; r < 8; ++r) {
      int m = mt * 16 + r + 8 * kh;
      if (m < N_TOK)
        out[(size_t)w * N_TOK * DIM_ + m * DIM_ + n] = c[r] + bias;
    }
  }
}

extern "C" void kernel_launch(void* const* d_in, const int* in_sizes, int n_in,
                              void* d_out, int out_size, void* d_ws, size_t ws_size,
                              hipStream_t stream) {
  const float* x          = (const float*)d_in[0];
  const float* mask       = (const float*)d_in[1];
  const float* w_qkv      = (const float*)d_in[2];
  const float* b_qkv      = (const float*)d_in[3];
  const float* bias_table = (const float*)d_in[4];
  const float* w_proj     = (const float*)d_in[5];
  const float* b_proj     = (const float*)d_in[6];
  const int*   rel_idx    = (const int*)d_in[7];
  float* out = (float*)d_out;

  // workspace layout: wq f16 frags (98304 B) | wp f16 frags (32768 B) | rel_bias f32 (38416 B)
  _Float16* wq = (_Float16*)d_ws;
  _Float16* wp = (_Float16*)((char*)d_ws + 98304);
  float*    rb = (float*)((char*)d_ws + 131072);

  prep_wq<<<192, 256, 0, stream>>>(w_qkv, wq);
  prep_wp<<<64, 256, 0, stream>>>(w_proj, wp);
  prep_rb<<<38, 256, 0, stream>>>(bias_table, rel_idx, rb);

  const size_t lds_bytes = 122880;  // 2 blocks per 320KB WGP
  swin_win_attn<<<4096, 256, lds_bytes, stream>>>(x, mask, b_qkv, b_proj, wq, wp, rb, out);
}
